// WavLMAttention_72164040507657
// MI455X (gfx1250) — matrix operand order repaired and checked by reading, not compiled, not run on hardware
//
#include <hip/hip_runtime.h>
#include <hip/hip_bf16.h>
#include <math.h>
#include <stdint.h>

// ---------------------------------------------------------------------------
// WavLM attention, fused for MI455X (gfx1250, wave32, WMMA, async-to-LDS, TDM).
// bf16 operands + fp32 WMMA accumulation (v_wmma_f32_16x16x32_bf16).
// Flash-attention fusion keeps the 256MB score tensor out of HBM/L2.
// GEMM + flash K/V tiles staged with GLOBAL_LOAD_ASYNC_TO_LDS_B128
// (ASYNCcnt, double-buffered, s_wait_asynccnt); per-head bias row staged with
// the Tensor Data Mover (tensor_load_to_lds, TENSORcnt, s_wait_tensorcnt).
// ---------------------------------------------------------------------------

typedef __attribute__((ext_vector_type(16))) __bf16 v16bf;
typedef __attribute__((ext_vector_type(8)))  float  v8f;
typedef unsigned int u32x4 __attribute__((ext_vector_type(4)));
typedef int          i32x8 __attribute__((ext_vector_type(8)));
typedef int          i32x4 __attribute__((ext_vector_type(4)));

#define B_  4
#define T_  1024
#define D_  1024
#define H_  16
#define HD_ 64

static __device__ __forceinline__ unsigned short f2bf(float f) {
  unsigned int x = __float_as_uint(f);
  x += 0x7fffu + ((x >> 16) & 1u);          // round-to-nearest-even
  return (unsigned short)(x >> 16);
}

static __device__ __forceinline__ v8f v8f_zero() {
  v8f z = {0.f,0.f,0.f,0.f,0.f,0.f,0.f,0.f};
  return z;
}

static __device__ __forceinline__ v8f wmma_bf16(v16bf a, v16bf b, v8f c) {
  return __builtin_amdgcn_wmma_f32_16x16x32_bf16(false, a, false, b,
                                                 (short)0, c, false, false);
}

// Generic (flat) pointer to a __shared__ object: low 32 bits are the LDS byte
// address.
static __device__ __forceinline__ unsigned lds_addr(const void* p) {
  return (unsigned)(uintptr_t)p;
}

// Async direct-to-LDS copies (per-lane 16B), tracked with ASYNCcnt.
// INST_OFFSET advances BOTH the LDS and global addresses (ISA 08, §4.4).
static __device__ __forceinline__ void async_copy_b128_x2(unsigned lds, const void* gaddr) {
  asm volatile(
      "global_load_async_to_lds_b128 %0, %1, off\n\t"
      "global_load_async_to_lds_b128 %0, %1, off offset:16"
      :: "v"(lds), "v"((unsigned long long)(uintptr_t)gaddr) : "memory");
}
static __device__ __forceinline__ void async_copy_b128_x4(unsigned lds, const void* gaddr) {
  asm volatile(
      "global_load_async_to_lds_b128 %0, %1, off\n\t"
      "global_load_async_to_lds_b128 %0, %1, off offset:16\n\t"
      "global_load_async_to_lds_b128 %0, %1, off offset:32\n\t"
      "global_load_async_to_lds_b128 %0, %1, off offset:48"
      :: "v"(lds), "v"((unsigned long long)(uintptr_t)gaddr) : "memory");
}
static __device__ __forceinline__ void wait_async_le6() {
  asm volatile("s_wait_asynccnt 0x6" ::: "memory");
}
static __device__ __forceinline__ void wait_async_le4() {
  asm volatile("s_wait_asynccnt 0x4" ::: "memory");
}
static __device__ __forceinline__ void wait_async_0() {
  asm volatile("s_wait_asynccnt 0x0" ::: "memory");
}

// TDM: load one contiguous row of `nelem` fp32 from global into LDS.
// D# per CDNA5 ISA: group0 = {count, lds_addr, global_addr[56:0], type=2},
// group1 = {data_size=4B, tensor_dim0/1, tile_dim0/1, dim0_stride},
// groups 2/3 zero (dims <= 2). Tracked with TENSORcnt.
// amdgpu-toolchain (clang-23) builtin: 6 args, trailing i32x8 group + cpol.
static __device__ __forceinline__ void tdm_load_row_f32(unsigned lds_off,
                                                        const void* gsrc,
                                                        unsigned nelem) {
  unsigned long long ga = (unsigned long long)(uintptr_t)gsrc;
  u32x4 g0;
  g0[0] = 1u;                                               // count=1, no gather
  g0[1] = lds_off;                                          // lds_addr
  g0[2] = (unsigned)(ga & 0xffffffffull);                   // global_addr[31:0]
  g0[3] = (unsigned)((ga >> 32) & 0x01ffffffull) | (2u << 30); // [56:32] | type=2
  i32x8 g1;
  g1[0] = (2 << 16);                                        // data_size=4B, mask=0
  g1[1] = (int)((nelem & 0xffffu) << 16);                   // tensor_dim0[15:0]
  g1[2] = (int)((1u << 16) | (nelem >> 16));                // tensor_dim1=1 | dim0 hi
  g1[3] = (int)((nelem & 0xffffu) << 16);                   // tile_dim0 = nelem
  g1[4] = 1;                                                // tile_dim1 = 1
  g1[5] = (int)nelem;                                       // tensor_dim0_stride
  g1[6] = 0;
  g1[7] = 0;
  i32x4 z4 = {0, 0, 0, 0};
  i32x8 z8 = {0, 0, 0, 0, 0, 0, 0, 0};
  __builtin_amdgcn_tensor_load_to_lds(g0, g1, z4, z4, z8, 0);
}

// A fragment: 16x32 bf16 tile, row-major source with row stride ldk (elems).
// ISA layout: lanes 0-15 hold K=[half? 8..15 : 0..7] then [24..31 : 16..23].
static __device__ __forceinline__ v16bf load_a_frag(const unsigned short* p0, int ldk) {
  const int lane = threadIdx.x & 31, half = lane >> 4, r = lane & 15;
  const unsigned short* p = p0 + (size_t)r * ldk;
  v16bf a;
  ((uint4*)&a)[0] = *(const uint4*)(p + half * 8);
  ((uint4*)&a)[1] = *(const uint4*)(p + 16 + half * 8);
  return a;
}

// B fragment: 32x16 bf16 tile fed from a row-major [N][K] source (i.e. B^T),
// column n = lane&15, reduction index kk = half*8 + e, then 16 + half*8 + e.
static __device__ __forceinline__ v16bf load_b_frag(const unsigned short* p0, int ldk) {
  const int lane = threadIdx.x & 31, half = lane >> 4, n = lane & 15;
  const unsigned short* p = p0 + (size_t)n * ldk + half * 8;
  v16bf b;
  ((uint4*)&b)[0] = *(const uint4*)(p);
  ((uint4*)&b)[1] = *(const uint4*)(p + 16);
  return b;
}

// ---------------------------------------------------------------------------
// fp32 -> bf16 conversion
// ---------------------------------------------------------------------------
__global__ void cvt_bf16_kernel(const float* __restrict__ in,
                                unsigned short* __restrict__ out, int n) {
  int i = blockIdx.x * blockDim.x + threadIdx.x;
  if (i < n) out[i] = f2bf(in[i]);
}

// ---------------------------------------------------------------------------
// Relative-position bias table: bucket() depends only on distance d=k-q, so
// precompute table_t[h][d+1023] = rel_embed[bucket(d)][h]  (16 x 2047,
// TRANSPOSED so the flash kernel TDM-loads one contiguous row per head).
// ---------------------------------------------------------------------------
__global__ void bias_table_kernel(const float* __restrict__ rel_embed,
                                  float* __restrict__ table_t) {
  int idx = blockIdx.x * blockDim.x + threadIdx.x;   // 2047*H
  if (idx >= 2047 * H_) return;
  int dd = idx / H_;
  int dpos = dd - 1023;
  int h = idx % H_;
  int rb = (dpos > 0) ? 160 : 0;
  int rp = (dpos < 0) ? -dpos : dpos;
  int val;
  if (rp < 80) {
    val = rp;
  } else {
    float l = __logf((float)rp * (1.0f / 80.0f)) * (80.0f / 2.302585092994046f);
    int lv = 80 + (int)l;
    val = lv < 159 ? lv : 159;
  }
  table_t[(size_t)h * 2047 + dd] = rel_embed[(rb + val) * H_ + h];
}

// ---------------------------------------------------------------------------
// GRU-style gate: per (b,h,t) scalar gate_out = ga*(gb*const - 1) + 2
// ---------------------------------------------------------------------------
__global__ void gates_kernel(const float* __restrict__ hidden,
                             const float* __restrict__ gru_w,
                             const float* __restrict__ gru_b,
                             const float* __restrict__ gru_const,
                             float* __restrict__ gate) {
  int idx = blockIdx.x * blockDim.x + threadIdx.x;   // B*H*T
  if (idx >= B_ * H_ * T_) return;
  int t = idx & (T_ - 1);
  int h = (idx >> 10) & (H_ - 1);
  int b = idx >> 14;
  const float* g = hidden + ((size_t)b * T_ + t) * D_ + h * HD_;
  float acc[8];
#pragma unroll
  for (int j = 0; j < 8; ++j) acc[j] = gru_b[j];
  for (int d = 0; d < HD_; ++d) {
    float gv = g[d];
#pragma unroll
    for (int j = 0; j < 8; ++j) acc[j] += gv * gru_w[j * HD_ + d];
  }
  float pa = acc[0] + acc[1] + acc[2] + acc[3];
  float pb = acc[4] + acc[5] + acc[6] + acc[7];
  float ga  = 1.f / (1.f + __expf(-pa));
  float gbv = 1.f / (1.f + __expf(-pb));
  gate[idx] = ga * (gbv * gru_const[h] - 1.0f) + 2.0f;
}

// ---------------------------------------------------------------------------
// bf16 GEMM: C[m,n] = sum_k A[m,k]*B[n,k] (B given transposed, K contiguous).
// Block tile 128x64, 8 waves, each wave 32x32. K-step 64 (8 WMMA / iter).
// Async-to-LDS double-buffered fill; last iteration peeled so the steady
// loop is branch-free: issue(next) -> s_wait_asynccnt 6 -> barrier -> WMMA.
// mode 0: bf16 out, [B,H,T,hd] head-major (Q/K), scaled
// mode 1: bf16 out, [B,H,hd,T] transposed (V) for contiguous PV fragments
// mode 2: fp32 out, row-major [M,N] (output projection -> d_out)
// ---------------------------------------------------------------------------
__global__ __launch_bounds__(256) void gemm_bf16_kernel(
    const unsigned short* __restrict__ A,   // [M][K] bf16
    const unsigned short* __restrict__ Bw,  // [N][K] bf16
    const float* __restrict__ bias,         // [N]
    float scale, int M, int N, int K, int mode,
    unsigned short* __restrict__ obf, float* __restrict__ of) {
  __shared__ __align__(16) unsigned short sA[2][128 * 64];
  __shared__ __align__(16) unsigned short sB[2][64 * 64];
  const int tid = threadIdx.x;
  const int m_blk = blockIdx.y * 128, n_blk = blockIdx.x * 64;
  const int wave = tid >> 5, lane = tid & 31, half = lane >> 4, l16 = lane & 15;
  const int wm = (wave & 3) * 32, wn = (wave >> 2) * 32;
  v8f c00 = v8f_zero(), c01 = v8f_zero(), c10 = v8f_zero(), c11 = v8f_zero();

  // Staging map: A 128x64 (64B/thread = 4x b128), B 64x64 (32B/thread = 2x).
  const int ar = tid >> 1, ac = (tid & 1) * 32;
  const int br = tid >> 2, bc = (tid & 3) * 16;
  const unsigned short* Ag = A  + (size_t)(m_blk + ar) * K + ac;
  const unsigned short* Bg = Bw + (size_t)(n_blk + br) * K + bc;

  auto issue_tile = [&](int k0, int buf) {
    async_copy_b128_x4(lds_addr(&sA[buf][ar * 64 + ac]), Ag + k0);
    async_copy_b128_x2(lds_addr(&sB[buf][br * 64 + bc]), Bg + k0);
  };
  auto compute_tile = [&](int buf) {
#pragma unroll
    for (int ch = 0; ch < 2; ++ch) {
      v16bf a0 = load_a_frag(&sA[buf][(wm +  0) * 64 + ch * 32], 64);
      v16bf a1 = load_a_frag(&sA[buf][(wm + 16) * 64 + ch * 32], 64);
      v16bf b0 = load_b_frag(&sB[buf][(wn +  0) * 64 + ch * 32], 64);
      v16bf b1 = load_b_frag(&sB[buf][(wn + 16) * 64 + ch * 32], 64);
      c00 = wmma_bf16(a0, b0, c00);
      c01 = wmma_bf16(a0, b1, c01);
      c10 = wmma_bf16(a1, b0, c10);
      c11 = wmma_bf16(a1, b1, c11);
    }
  };

  issue_tile(0, 0);
  int buf = 0;
  for (int k0 = 0; k0 + 64 < K; k0 += 64) {     // steady state, branch-free
    issue_tile(k0 + 64, buf ^ 1);
    wait_async_le6();
    __syncthreads();
    compute_tile(buf);
    __syncthreads();   // all waves done reading buf before it is refilled
    buf ^= 1;
  }
  wait_async_0();                                // peeled last tile
  __syncthreads();
  compute_tile(buf);

  v8f* cs[4] = {&c00, &c01, &c10, &c11};
#pragma unroll
  for (int i = 0; i < 2; ++i) {
#pragma unroll
    for (int j = 0; j < 2; ++j) {
      v8f cv = *cs[i * 2 + j];
#pragma unroll
      for (int r = 0; r < 8; ++r) {
        int m = m_blk + wm + i * 16 + r + half * 8;   // C row: M=r / M=r+8
        int n = n_blk + wn + j * 16 + l16;            // C col across lanes
        float v = (cv[r] + bias[n]) * scale;
        if (mode == 2) {
          of[(size_t)m * N + n] = v;
        } else {
          int b = m >> 10, t = m & (T_ - 1);   // M = B*T, T = 1024
          int h = n >> 6,  d = n & (HD_ - 1);  // N = D, hd = 64
          size_t idx;
          if (mode == 0) idx = (((size_t)b * H_ + h) * T_  + t) * HD_ + d;
          else           idx = (((size_t)b * H_ + h) * HD_ + d) * T_  + t;
          obf[idx] = f2bf(v);
        }
      }
    }
  }
}

// ---------------------------------------------------------------------------
// Flash attention per (b,h): 128 threads = 4 waves, each wave owns 16 query
// rows; 32-key tiles. K/V tiles async-staged into LDS once per block
// (double-buffered, shared by all 4 waves). The per-head bias row (2047 fp32)
// is staged by the Tensor Data Mover (one descriptor, wave 0, TENSORcnt).
// ---------------------------------------------------------------------------
__global__ __launch_bounds__(128) void flash_attn_kernel(
    const unsigned short* __restrict__ Q,    // [B,H,T,hd] bf16
    const unsigned short* __restrict__ Km,   // [B,H,T,hd] bf16
    const unsigned short* __restrict__ Vt,   // [B,H,hd,T] bf16 (transposed)
    const float* __restrict__ table_t,       // [H][2047]
    const float* __restrict__ gate,          // [B,H,T]
    unsigned short* __restrict__ ctx) {      // [B,T,H,hd] bf16
  __shared__ __align__(16) float tbl[2047];
  __shared__ __align__(16) unsigned short sK[2][32 * 64];  // [key][d]
  __shared__ __align__(16) unsigned short sV[2][64 * 32];  // [d][key]
  __shared__ __align__(16) unsigned short sP[4 * 16 * 32]; // wave-private P
  const int b = blockIdx.z, h = blockIdx.y, q0 = blockIdx.x * 64;
  const int tid = threadIdx.x;
  const int wave = tid >> 5, lane = tid & 31, half = lane >> 4, l16 = lane & 15;
  const int qw = q0 + wave * 16;
  const unsigned short* Qb = Q  + (((size_t)b * H_ + h) * T_) * HD_;
  const unsigned short* Kb = Km + (((size_t)b * H_ + h) * T_) * HD_;
  const unsigned short* Vb = Vt + (((size_t)b * H_ + h) * HD_) * T_;
  const float* gb = gate + ((size_t)b * H_ + h) * T_;
  unsigned short* sPw = sP + wave * 16 * 32;

  // K tile: 32 rows x 128B; V tile: 64 rows x 64B  (32B/thread each).
  const int kr = tid >> 2, kc = (tid & 3) * 16;    // K: row 0..31, col elems
  const int vr = tid >> 1, vc = (tid & 1) * 16;    // V: row 0..63, col elems
  auto issue_kv = [&](int kt, int buf) {
    async_copy_b128_x2(lds_addr(&sK[buf][kr * 64 + kc]), Kb + (size_t)(kt + kr) * HD_ + kc);
    async_copy_b128_x2(lds_addr(&sV[buf][vr * 32 + vc]), Vb + (size_t)vr * T_ + kt + vc);
  };

  issue_kv(0, 0);
  if (wave == 0) {   // TDM: contiguous 2047-float bias row -> LDS
    tdm_load_row_f32(lds_addr(&tbl[0]), table_t + (size_t)h * 2047, 2047u);
    __builtin_amdgcn_s_wait_tensorcnt(0);
  }

  // Q fragments for this wave's 16 rows, K-dim 64 split into two 32-chunks.
  v16bf aq0 = load_a_frag(Qb + (size_t)qw * HD_,      HD_);
  v16bf aq1 = load_a_frag(Qb + (size_t)qw * HD_ + 32, HD_);

  float gate_r[8], m_run[8], l_run[8];
#pragma unroll
  for (int r = 0; r < 8; ++r) {
    gate_r[r] = gb[qw + r + half * 8];
    m_run[r] = -1e30f;
    l_run[r] = 0.f;
  }
  v8f o0 = v8f_zero(), o1 = v8f_zero(), o2 = v8f_zero(), o3 = v8f_zero();

  auto process = [&](int kt, int buf) {
    // S(16x32) = Q(16x64) @ K^T : 4 WMMA (2 key subtiles x 2 K-chunks)
    v8f s0 = v8f_zero(), s1 = v8f_zero();
    s0 = wmma_bf16(aq0, load_b_frag(&sK[buf][ 0 * 64 +  0], 64), s0);
    s0 = wmma_bf16(aq1, load_b_frag(&sK[buf][ 0 * 64 + 32], 64), s0);
    s1 = wmma_bf16(aq0, load_b_frag(&sK[buf][16 * 64 +  0], 64), s1);
    s1 = wmma_bf16(aq1, load_b_frag(&sK[buf][16 * 64 + 32], 64), s1);

    // gated bias + online softmax (row reductions across 16-lane halves)
    float p0a[8], p1a[8];
#pragma unroll
    for (int r = 0; r < 8; ++r) {
      const int qrow = qw + r + half * 8;
      float v0 = s0[r] + gate_r[r] * tbl[kt + l16      - qrow + 1023];
      float v1 = s1[r] + gate_r[r] * tbl[kt + 16 + l16 - qrow + 1023];
      float mx = fmaxf(fmaxf(v0, v1), m_run[r]);
#pragma unroll
      for (int off = 1; off < 16; off <<= 1) mx = fmaxf(mx, __shfl_xor(mx, off, 16));
      float alpha = __expf(m_run[r] - mx);
      float e0 = __expf(v0 - mx), e1 = __expf(v1 - mx);
      float rs = e0 + e1;
#pragma unroll
      for (int off = 1; off < 16; off <<= 1) rs += __shfl_xor(rs, off, 16);
      l_run[r] = l_run[r] * alpha + rs;
      m_run[r] = mx;
      o0[r] *= alpha; o1[r] *= alpha; o2[r] *= alpha; o3[r] *= alpha;
      p0a[r] = e0; p1a[r] = e1;
    }

    // Repack P (C layout, fp32) -> A layout bf16 via wave-private LDS bounce.
#pragma unroll
    for (int r = 0; r < 8; ++r) {
      sPw[(r + half * 8) * 32 + l16]      = f2bf(p0a[r]);
      sPw[(r + half * 8) * 32 + 16 + l16] = f2bf(p1a[r]);
    }
    asm volatile("" ::: "memory");
    v16bf ap = load_a_frag(sPw, 32);

    // O(16x64) += P(16x32) @ V(32x64): 4 WMMA from the shared V tile.
    o0 = wmma_bf16(ap, load_b_frag(&sV[buf][ 0 * 32], 32), o0);
    o1 = wmma_bf16(ap, load_b_frag(&sV[buf][16 * 32], 32), o1);
    o2 = wmma_bf16(ap, load_b_frag(&sV[buf][32 * 32], 32), o2);
    o3 = wmma_bf16(ap, load_b_frag(&sV[buf][48 * 32], 32), o3);
  };

  int buf = 0;
  for (int kt = 0; kt + 32 < T_; kt += 32) {     // steady state
    issue_kv(kt + 32, buf ^ 1);
    wait_async_le4();
    __syncthreads();
    process(kt, buf);
    __syncthreads();   // everyone done with buf before refill
    buf ^= 1;
  }
  wait_async_0();                                 // peeled last tile
  __syncthreads();
  process(T_ - 32, buf);

  // Normalize and write ctx back in [B,T,H,hd] so out-proj GEMM reads [M,K].
  v8f* os[4] = {&o0, &o1, &o2, &o3};
#pragma unroll
  for (int j = 0; j < 4; ++j) {
    v8f ov = *os[j];
#pragma unroll
    for (int r = 0; r < 8; ++r) {
      const int qrow = qw + r + half * 8;
      const int d = j * 16 + l16;
      float v = ov[r] / l_run[r];
      ctx[(((size_t)b * T_ + qrow) * H_ + h) * HD_ + d] = f2bf(v);
    }
  }
}

// ---------------------------------------------------------------------------
extern "C" void kernel_launch(void* const* d_in, const int* in_sizes, int n_in,
                              void* d_out, int out_size, void* d_ws, size_t ws_size,
                              hipStream_t stream) {
  (void)in_sizes; (void)n_in; (void)out_size; (void)ws_size;
  const float* hidden    = (const float*)d_in[0];
  const float* q_w       = (const float*)d_in[1];
  const float* q_b       = (const float*)d_in[2];
  const float* k_w       = (const float*)d_in[3];
  const float* k_b       = (const float*)d_in[4];
  const float* v_w       = (const float*)d_in[5];
  const float* v_b       = (const float*)d_in[6];
  const float* out_w     = (const float*)d_in[7];
  const float* out_b     = (const float*)d_in[8];
  const float* rel_embed = (const float*)d_in[9];
  const float* gru_const = (const float*)d_in[10];
  const float* gru_w     = (const float*)d_in[11];
  const float* gru_b     = (const float*)d_in[12];
  float* out = (float*)d_out;

  char* ws = (char*)d_ws;
  size_t off = 0;
  auto take = [&](size_t bytes) -> char* {
    char* p = ws + off;
    off = (off + bytes + 255) & ~(size_t)255;
    return p;
  };
  const size_t NACT = (size_t)B_ * T_ * D_;   // 4M elems
  const size_t NW   = (size_t)D_ * D_;        // 1M elems
  unsigned short* hbf  = (unsigned short*)take(NACT * 2);
  unsigned short* wqb  = (unsigned short*)take(NW * 2);
  unsigned short* wkb  = (unsigned short*)take(NW * 2);
  unsigned short* wvb  = (unsigned short*)take(NW * 2);
  unsigned short* wob  = (unsigned short*)take(NW * 2);
  unsigned short* qbf  = (unsigned short*)take(NACT * 2);
  unsigned short* kbf  = (unsigned short*)take(NACT * 2);
  unsigned short* vtb  = (unsigned short*)take(NACT * 2);
  unsigned short* ctxb = (unsigned short*)take(NACT * 2);
  float* tblT = (float*)take((size_t)H_ * 2047 * 4);
  float* gate = (float*)take((size_t)B_ * H_ * T_ * 4);

  // bf16 conversions
  cvt_bf16_kernel<<<(int)((NACT + 255) / 256), 256, 0, stream>>>(hidden, hbf, (int)NACT);
  cvt_bf16_kernel<<<(int)((NW + 255) / 256), 256, 0, stream>>>(q_w,   wqb, (int)NW);
  cvt_bf16_kernel<<<(int)((NW + 255) / 256), 256, 0, stream>>>(k_w,   wkb, (int)NW);
  cvt_bf16_kernel<<<(int)((NW + 255) / 256), 256, 0, stream>>>(v_w,   wvb, (int)NW);
  cvt_bf16_kernel<<<(int)((NW + 255) / 256), 256, 0, stream>>>(out_w, wob, (int)NW);

  // bias table (transposed for TDM row loads) + gates
  bias_table_kernel<<<(2047 * H_ + 255) / 256, 256, 0, stream>>>(rel_embed, tblT);
  gates_kernel<<<(B_ * H_ * T_ + 255) / 256, 256, 0, stream>>>(hidden, gru_w, gru_b,
                                                               gru_const, gate);

  // Q / K / V projections (M=4096, N=1024, K=1024)
  dim3 gg(D_ / 64, (B_ * T_) / 128);
  gemm_bf16_kernel<<<gg, 256, 0, stream>>>(hbf, wqb, q_b, 0.125f /* hd^-0.5 */,
                                           B_ * T_, D_, D_, 0, qbf, nullptr);
  gemm_bf16_kernel<<<gg, 256, 0, stream>>>(hbf, wkb, k_b, 1.0f,
                                           B_ * T_, D_, D_, 0, kbf, nullptr);
  gemm_bf16_kernel<<<gg, 256, 0, stream>>>(hbf, wvb, v_b, 1.0f,
                                           B_ * T_, D_, D_, 1, vtb, nullptr);

  // Fused flash attention with gated relative-position bias
  flash_attn_kernel<<<dim3(T_ / 64, H_, B_), 128, 0, stream>>>(qbf, kbf, vtb,
                                                               tblT, gate, ctxb);

  // Output projection -> fp32 d_out
  gemm_bf16_kernel<<<gg, 256, 0, stream>>>(ctxb, wob, out_b, 1.0f,
                                           B_ * T_, D_, D_, 2, nullptr, out);
}
